// QWen3Attention_54855322304982
// MI455X (gfx1250) — matrix operand order repaired and checked by reading, not compiled, not run on hardware
//
#include <hip/hip_runtime.h>
#include <hip/hip_bf16.h>
#include <cstdint>

// ---------------------------------------------------------------------------
// Qwen3 attention layer for MI455X (gfx1250, wave32, WMMA + TDM + async-LDS).
// GEMMs (wq/wk/wv, q@k^T, p@v, wo) on v_wmma_f32_16x16x32_f16 (fp32 accum).
// A-tiles staged via global_load_async_to_lds_b128 (ASYNCcnt), attention
// K-tiles via tensor_load_to_lds (TENSORcnt), softmax/RMSNorm/RoPE fp32 VALU.
// ---------------------------------------------------------------------------

typedef __attribute__((ext_vector_type(16))) _Float16     v16h;
typedef __attribute__((ext_vector_type(8)))  float        v8f;
typedef __attribute__((ext_vector_type(4)))  unsigned int u32x4;
typedef __attribute__((ext_vector_type(8)))  unsigned int u32x8;

// ---- WMMA fragment loaders (ISA 7.12.2 layouts, 16-bit data, wave32) ------
// A-matrix 16x32: lane m=lane&15, half g=lane>>4; VGPR j holds
// K = (j/4)*16 + g*8 + (j%4)*2 (+1).
__device__ __forceinline__ v16h load_a_frag(const _Float16* base, int stride) {
  int lane = threadIdx.x & 31;
  int m = lane & 15, g = lane >> 4;
  const _Float16* row = base + m * stride;
  v16h f;
#pragma unroll
  for (int j = 0; j < 8; ++j) {
    int k0 = ((j >> 2) << 4) + (g << 3) + ((j & 3) << 1);
    f[2 * j]     = row[k0];
    f[2 * j + 1] = row[k0 + 1];
  }
  return f;
}

// B-matrix 32x16, given as B^T rows (n-major [16][K]): lane n=lane&15,
// g=lane>>4; VGPR j holds K = (j/4)*16 + g*8 + (j%4)*2 (+1)  (same as A).
__device__ __forceinline__ v16h load_b_frag(const _Float16* base, int stride) {
  int lane = threadIdx.x & 31;
  int n = lane & 15, g = lane >> 4;
  const _Float16* row = base + n * stride;
  v16h f;
#pragma unroll
  for (int j = 0; j < 8; ++j) {
    int k0 = ((j >> 2) << 4) + (g << 3) + ((j & 3) << 1);
    f[2 * j]     = row[k0];
    f[2 * j + 1] = row[k0 + 1];
  }
  return f;
}

// ---- async global -> LDS 16B copy (one lane = one b128 transfer) ----------
__device__ __forceinline__ void async_copy_b128(const _Float16* gsrc, _Float16* ldst) {
  unsigned lo  = (unsigned)(uintptr_t)ldst;       // wave-relative LDS offset
  uint64_t ga  = (uint64_t)(uintptr_t)gsrc;
  asm volatile("global_load_async_to_lds_b128 %0, %1, off"
               :: "v"(lo), "v"(ga) : "memory");
}
__device__ __forceinline__ void wait_async() {
  asm volatile("s_wait_asynccnt 0x0" ::: "memory");
}

// ---- elementwise fp32 -> f16 cast -----------------------------------------
__global__ void castk(const float* __restrict__ s, _Float16* __restrict__ d, int n) {
  int i = blockIdx.x * 256 + threadIdx.x;
  if (i < n) d[i] = (_Float16)s[i];
}

// V: [token][kvh*128+d] fp32 -> [b][kvh][s][128] f16
__global__ void vcast(const float* __restrict__ s, _Float16* __restrict__ d) {
  int i = blockIdx.x * 256 + threadIdx.x;           // 4096*1024 elements
  int t = i >> 10, c = i & 1023;
  int h = c >> 7, dd = c & 127;
  int b = t >> 10, sq = t & 1023;
  d[(((size_t)(b * 8 + h) * 1024) + sq) * 128 + dd] = (_Float16)s[i];
}

// ---- tiled WMMA GEMM: C[M,N] = A[M,K] * B[K,N], f16 in, fp32 out ----------
#define LDA_S 40
#define LDB_S 40
__global__ __launch_bounds__(256) void gemm_f16f32(
    const _Float16* __restrict__ A, const _Float16* __restrict__ B,
    float* __restrict__ C, int M, int N, int K) {
  __shared__ _Float16 lA[128 * LDA_S];      // [m][k] row-major
  __shared__ _Float16 lB[64 * LDB_S];       // transposed: [n][k]
  int tid = threadIdx.x;
  int w = tid >> 5, lane = tid & 31;
  int m0 = blockIdx.y * 128, n0 = blockIdx.x * 64;
  v8f acc[4] = {};
  for (int k0 = 0; k0 < K; k0 += 32) {
#pragma unroll
    for (int i = 0; i < 2; ++i) {           // A tile 128x32: async to LDS
      int v = tid + i * 256;
      int r = v >> 2, c = (v & 3) << 3;
      async_copy_b128(A + (size_t)(m0 + r) * K + k0 + c, &lA[r * LDA_S + c]);
    }
    {                                        // B tile 32x64, store transposed
      int kk = tid >> 3, nv = (tid & 7) << 3;
      uint4 dv = *(const uint4*)(B + (size_t)(k0 + kk) * N + n0 + nv);
      _Float16 t[8]; *(uint4*)t = dv;
#pragma unroll
      for (int j = 0; j < 8; ++j) lB[(nv + j) * LDB_S + kk] = t[j];
      if (k0 + 32 < K)                       // global_prefetch_b8 on next B tile
        __builtin_prefetch(B + (size_t)(k0 + 32 + kk) * N + n0 + nv, 0, 1);
    }
    wait_async();
    __syncthreads();
    v16h a = load_a_frag(&lA[w * 16 * LDA_S], LDA_S);
#pragma unroll
    for (int ns = 0; ns < 4; ++ns) {
      v16h bf = load_b_frag(&lB[ns * 16 * LDB_S], LDB_S);
      acc[ns] = __builtin_amdgcn_wmma_f32_16x16x32_f16(
          false, a, false, bf, (short)0, acc[ns], false, false);
    }
    __syncthreads();
  }
  int g = lane >> 4, nn = lane & 15;
#pragma unroll
  for (int ns = 0; ns < 4; ++ns)
#pragma unroll
    for (int r = 0; r < 8; ++r) {
      int m = m0 + w * 16 + r + 8 * g;
      C[(size_t)m * N + n0 + ns * 16 + nn] = acc[ns][r];
    }
}

// ---- per-head RMSNorm + interleaved RoPE, fp32 in, f16 [B][H][S][128] out -
__global__ __launch_bounds__(256) void normrope(
    const float* __restrict__ src, const float* __restrict__ scale,
    const int* __restrict__ pos, _Float16* __restrict__ dst, int nheads) {
  int warp = threadIdx.x >> 5, lane = threadIdx.x & 31;
  int idx = blockIdx.x * 8 + warp;          // token*nheads + h
  int t = idx / nheads, h = idx - t * nheads;
  int b = t >> 10, sq = t & 1023;
  const float* x = src + (size_t)t * (nheads * 128) + h * 128;
  int d0 = lane * 4;
  float x0 = x[d0], x1 = x[d0 + 1], x2 = x[d0 + 2], x3 = x[d0 + 3];
  float ss = x0 * x0 + x1 * x1 + x2 * x2 + x3 * x3;
#pragma unroll
  for (int mask = 16; mask; mask >>= 1) ss += __shfl_xor(ss, mask, 32);
  float r = rsqrtf(ss * (1.f / 128.f) + 1e-6f);
  x0 *= r * scale[d0];     x1 *= r * scale[d0 + 1];
  x2 *= r * scale[d0 + 2]; x3 *= r * scale[d0 + 3];
  float p = (float)pos[t];
  float f0 = p * __powf(1.0e6f, -(float)d0 * (1.f / 128.f));
  float f1 = p * __powf(1.0e6f, -(float)(d0 + 2) * (1.f / 128.f));
  float c0, s0, c1, s1;
  __sincosf(f0, &s0, &c0);
  __sincosf(f1, &s1, &c1);
  float o0 = x0 * c0 - x1 * s0, o1 = x1 * c0 + x0 * s0;
  float o2 = x2 * c1 - x3 * s1, o3 = x3 * c1 + x2 * s1;
  _Float16* o = dst + (((size_t)(b * nheads + h) * 1024) + sq) * 128 + d0;
  o[0] = (_Float16)o0; o[1] = (_Float16)o1;
  o[2] = (_Float16)o2; o[3] = (_Float16)o3;
}

// ---- causal GQA flash attention, 128-query tile per block -----------------
// K tiles arrive via the Tensor Data Mover; V tiles staged transposed by VALU.
#define LK_S 128
#define LV_S 36
#define LP_S 36
__global__ __launch_bounds__(256) void flash_attn(
    const _Float16* __restrict__ Q,   // [B][32][S][128]
    const _Float16* __restrict__ Km,  // [B][8][S][128]
    const _Float16* __restrict__ Vm,  // [B][8][S][128]
    _Float16* __restrict__ attn) {    // [B*S][4096]
  __shared__ _Float16 lK[32 * LK_S];        // [key][d]  (TDM destination)
  __shared__ _Float16 lVT[128 * LV_S];      // [d][key]
  __shared__ _Float16 lP[8 * 16 * LP_S];    // per-wave 16x32 P staging
  int tid = threadIdx.x, w = tid >> 5, lane = tid & 31;
  int g = lane >> 4, ln = lane & 15;
  int qt = blockIdx.x, h = blockIdx.y, b = blockIdx.z;
  int kvh = h >> 2;
  const _Float16* Qb = Q + (((size_t)(b * 32 + h) * 1024) + qt * 128 + w * 16) * 128;
  const _Float16* Kb = Km + ((size_t)(b * 8 + kvh) * 1024) * 128;
  const _Float16* Vb = Vm + ((size_t)(b * 8 + kvh) * 1024) * 128;
  v16h qfrag[4];
#pragma unroll
  for (int c = 0; c < 4; ++c) qfrag[c] = load_a_frag(Qb + c * 32, 128);
  float m_i[8], l_i[8];
  v8f O[8] = {};
#pragma unroll
  for (int r = 0; r < 8; ++r) { m_i[r] = -3.0e30f; l_i[r] = 0.f; }
  int rowmax_wave = qt * 128 + w * 16 + 15;
  int nkt = (qt + 1) * 4;                   // key tiles of 32 up to (qt+1)*128
  for (int kt = 0; kt < nkt; ++kt) {
    if (w == 0) {
      // TDM: 2D tile 128x32 halfs, contiguous rows, into lK (cdna5 §8 D#)
      uint64_t ga = (uint64_t)(uintptr_t)(Kb + (size_t)(kt * 32) * 128);
      u32x4 g0;
      g0[0] = 1u;                                    // count=1, user mode
      g0[1] = (unsigned)(uintptr_t)&lK[0];           // lds_addr
      g0[2] = (unsigned)ga;                          // global_addr[31:0]
      g0[3] = ((unsigned)(ga >> 32) & 0x01FFFFFFu)   // global_addr[56:32]
              | 0x80000000u;                         // type=2 ("image")
      u32x8 g1;
      g1[0] = 0x00010000u;                           // data_size=1 (2 bytes)
      g1[1] = (128u & 0xffffu) << 16;                // tensor_dim0=128 (lo16)
      g1[2] = 0u | ((32u & 0xffffu) << 16);          // td0 hi16, tensor_dim1=32
      g1[3] = 0u | (128u << 16);                     // td1 hi16, tile_dim0=128
      g1[4] = 32u;                                   // tile_dim1=32, tile_dim2=0
      g1[5] = 128u;                                  // tensor_dim0_stride lo32
      g1[6] = 0u;                                    // stride0 hi, stride1 lo
      g1[7] = 0u;
      u32x4 gz = {0u, 0u, 0u, 0u};
      asm volatile("tensor_load_to_lds %0, %1, %2, %3"
                   :: "s"(g0), "s"(g1), "s"(gz), "s"(gz) : "memory");
      __builtin_amdgcn_s_wait_tensorcnt(0);
    }
#pragma unroll
    for (int i = 0; i < 2; ++i) {           // V tile, transposed into lVT
      int v = tid + i * 256;
      int r = v >> 4, cv = (v & 15) << 3;
      uint4 dv = *(const uint4*)(Vb + (size_t)(kt * 32 + r) * 128 + cv);
      _Float16 tv[8]; *(uint4*)tv = dv;
#pragma unroll
      for (int j = 0; j < 8; ++j) lVT[(cv + j) * LV_S + r] = tv[j];
    }
    __syncthreads();
    if (kt * 32 <= rowmax_wave) {
      float p[2][8];
#pragma unroll
      for (int ns = 0; ns < 2; ++ns) {      // S = Q * K^T  (16x16 per WMMA)
        v8f sc = {};
#pragma unroll
        for (int c = 0; c < 4; ++c) {
          v16h bf = load_b_frag(&lK[ns * 16 * LK_S + c * 32], LK_S);
          sc = __builtin_amdgcn_wmma_f32_16x16x32_f16(
              false, qfrag[c], false, bf, (short)0, sc, false, false);
        }
        int n = kt * 32 + ns * 16 + ln;
#pragma unroll
        for (int r = 0; r < 8; ++r) {
          int m = qt * 128 + w * 16 + r + 8 * g;
          float s = sc[r] * 0.0883883476483f;   // 1/sqrt(128)
          p[ns][r] = (n > m) ? -3.0e30f : s;
        }
      }
#pragma unroll
      for (int r = 0; r < 8; ++r) {         // online softmax per row
        float mx = fmaxf(p[0][r], p[1][r]);
#pragma unroll
        for (int mask = 8; mask; mask >>= 1) mx = fmaxf(mx, __shfl_xor(mx, mask, 32));
        float mnew = fmaxf(m_i[r], mx);
        float alpha = __expf(m_i[r] - mnew);
        float p0 = (p[0][r] < -1.0e30f) ? 0.f : __expf(p[0][r] - mnew);
        float p1 = (p[1][r] < -1.0e30f) ? 0.f : __expf(p[1][r] - mnew);
        float sum = p0 + p1;
#pragma unroll
        for (int mask = 8; mask; mask >>= 1) sum += __shfl_xor(sum, mask, 32);
        l_i[r] = l_i[r] * alpha + sum;
        m_i[r] = mnew;
#pragma unroll
        for (int t2 = 0; t2 < 8; ++t2) O[t2][r] *= alpha;
        int row = r + 8 * g;
        lP[(w * 16 + row) * LP_S + ln]      = (_Float16)p0;
        lP[(w * 16 + row) * LP_S + 16 + ln] = (_Float16)p1;
      }
      v16h pf = load_a_frag(&lP[w * 16 * LP_S], LP_S);
#pragma unroll
      for (int dsb = 0; dsb < 8; ++dsb) {   // O += P * V
        v16h vf2 = load_b_frag(&lVT[dsb * 16 * LV_S], LV_S);
        O[dsb] = __builtin_amdgcn_wmma_f32_16x16x32_f16(
            false, pf, false, vf2, (short)0, O[dsb], false, false);
      }
    }
    __syncthreads();
  }
#pragma unroll
  for (int dsb = 0; dsb < 8; ++dsb)
#pragma unroll
    for (int r = 0; r < 8; ++r) {
      int srow = qt * 128 + w * 16 + r + 8 * g;
      float o = O[dsb][r] / l_i[r];
      attn[((size_t)(b * 1024 + srow)) * 4096 + h * 128 + dsb * 16 + ln] = (_Float16)o;
    }
}

// ---------------------------------------------------------------------------
extern "C" void kernel_launch(void* const* d_in, const int* in_sizes, int n_in,
                              void* d_out, int out_size, void* d_ws, size_t ws_size,
                              hipStream_t stream) {
  (void)in_sizes; (void)n_in; (void)out_size; (void)ws_size;
  const float* x   = (const float*)d_in[0];
  const int*   pos = (const int*)d_in[1];
  const float* wq  = (const float*)d_in[2];
  const float* wk  = (const float*)d_in[3];
  const float* wv  = (const float*)d_in[4];
  const float* wo  = (const float*)d_in[5];
  const float* qsc = (const float*)d_in[6];
  const float* ksc = (const float*)d_in[7];

  char* ws = (char*)d_ws;
  size_t off = 0;
  auto alloc = [&](size_t bytes) {
    char* p = ws + off;
    off = (off + bytes + 255) & ~(size_t)255;
    return p;
  };
  const size_t TOK = 4096;                  // B*S
  _Float16* xh   = (_Float16*)alloc(TOK * 4096 * 2);
  _Float16* wqh  = (_Float16*)alloc(4096 * 4096 * 2);
  _Float16* wkh  = (_Float16*)alloc(4096 * 1024 * 2);
  _Float16* wvh  = (_Float16*)alloc(4096 * 1024 * 2);
  _Float16* woh  = (_Float16*)alloc(4096 * 4096 * 2);
  float*    qf32 = (float*)alloc(TOK * 4096 * 4);
  float*    kf32 = (float*)alloc(TOK * 1024 * 4);
  float*    vf32 = (float*)alloc(TOK * 1024 * 4);
  _Float16* qh   = (_Float16*)alloc(TOK * 4096 * 2);
  _Float16* kh   = (_Float16*)alloc(TOK * 1024 * 2);
  _Float16* vh   = (_Float16*)alloc(TOK * 1024 * 2);
  _Float16* atth = (_Float16*)qf32;         // reuse: q_f32 dead after normrope

  // 1) casts
  castk<<<dim3(65536), 256, 0, stream>>>(x,  xh,  16777216);
  castk<<<dim3(65536), 256, 0, stream>>>(wq, wqh, 16777216);
  castk<<<dim3(16384), 256, 0, stream>>>(wk, wkh, 4194304);
  castk<<<dim3(16384), 256, 0, stream>>>(wv, wvh, 4194304);
  castk<<<dim3(65536), 256, 0, stream>>>(wo, woh, 16777216);

  // 2) QKV projections (WMMA)
  gemm_f16f32<<<dim3(64, 32), 256, 0, stream>>>(xh, wqh, qf32, 4096, 4096, 4096);
  gemm_f16f32<<<dim3(16, 32), 256, 0, stream>>>(xh, wkh, kf32, 4096, 1024, 4096);
  gemm_f16f32<<<dim3(16, 32), 256, 0, stream>>>(xh, wvh, vf32, 4096, 1024, 4096);

  // 3) RMSNorm + RoPE (q,k) and reshape/cast (v)
  normrope<<<dim3(4096 * 32 / 8), 256, 0, stream>>>(qf32, qsc, pos, qh, 32);
  normrope<<<dim3(4096 * 8 / 8),  256, 0, stream>>>(kf32, ksc, pos, kh, 8);
  vcast<<<dim3(4194304 / 256), 256, 0, stream>>>(vf32, vh);

  // 4) causal GQA flash attention (WMMA + TDM)
  flash_attn<<<dim3(8, 32, 4), 256, 0, stream>>>(qh, kh, vh, atth);

  // 5) output projection (WMMA) -> fp32 d_out
  gemm_f16f32<<<dim3(64, 32), 256, 0, stream>>>(atth, woh, (float*)d_out, 4096, 4096, 4096);
}
